// MaritimeGNN_74869869904134
// MI455X (gfx1250) — matrix operand order repaired and checked by reading, not compiled, not run on hardware
//
#include <hip/hip_runtime.h>
#include <hip/hip_bf16.h>
#include <math.h>

// ---------------- WMMA fragment helpers (CDNA5 16x16x32 f16) ----------------
typedef __attribute__((ext_vector_type(16))) _Float16 v16h;
typedef __attribute__((ext_vector_type(2)))  __fp16   fp16x2;   // matches cvt_pkrtz return
typedef __attribute__((ext_vector_type(8)))  float    v8f;

union FragU { v16h h; unsigned int u[8]; };

__device__ __forceinline__ unsigned int pk2(float a, float b) {
    union { fp16x2 h; unsigned int u; } c;
    c.h = __builtin_amdgcn_cvt_pkrtz(a, b);   // v_cvt_pk_rtz_f16_f32
    return c.u;
}

// A tile in LDS: row-major [16 rows][stride halves], row m = lane&15.
__device__ __forceinline__ v16h load_a_frag(const _Float16* lds, int stride) {
    int lane = threadIdx.x & 31;
    int m    = lane & 15;
    int hi8  = (lane >> 4) << 3;
    FragU f;
#pragma unroll
    for (int v = 0; v < 8; ++v) {
        int kb = (v < 4) ? (hi8 + 2 * v) : (16 + hi8 + 2 * (v - 4));
        f.u[v] = *(const unsigned int*)(lds + m * stride + kb);
    }
    return f.h;
}

// B tile in LDS stored transposed: [16 cols][stride halves of K], col n = lane&15.
__device__ __forceinline__ v16h load_b_frag(const _Float16* lds, int stride) {
    int lane = threadIdx.x & 31;
    int n    = lane & 15;
    int koff = (lane >> 4) << 3;
    FragU f;
#pragma unroll
    for (int v = 0; v < 8; ++v) {
        f.u[v] = *(const unsigned int*)(lds + n * stride + ((v < 4) ? (koff + 2 * v) : (16 + koff + 2 * (v - 4))));
    }
    return f.h;
}

__device__ __forceinline__ v8f wmma_f16(v16h a, v16h b, v8f c) {
    return __builtin_amdgcn_wmma_f32_16x16x32_f16(false, a, false, b, (short)0, c, false, false);
}

__device__ __forceinline__ void atomicMaxF(float* addr, float val) {
    if (val >= 0.0f) atomicMax((int*)addr, __float_as_int(val));
    else             atomicMin((unsigned int*)addr, __float_as_uint(val));
}

// ---------------- WMMA GEMM: C[M,N] = A[M,K] * B[K,N] ----------------
// One wave per block; each wave computes a 16 x (16*NT) strip of C.
// A fragment built directly from global (float4 + cvt_pk_rtz), B staged in LDS.
// Requires: K % 8 == 0, N-cols covered exactly by grid (HC multiple of 16*NT).
template <int NT>
__global__ void k_gemm_wmma(const float* __restrict__ A, const float* __restrict__ B,
                            float* __restrict__ C, int M, int K, int N) {
    __shared__ _Float16 sB[NT * 16 * 32];    // per tile t: [col n][k] transposed
    int m0 = blockIdx.x * 16;
    int n0 = blockIdx.y * (16 * NT);
    int lane = threadIdx.x;                  // blockDim.x == 32
    int m = lane & 15, hi8 = (lane >> 4) << 3;

    int mrow = m0 + m; if (mrow >= M) mrow = M - 1;       // clamp read, store guarded
    const float* arow = A + (size_t)mrow * K;

    v8f acc[NT];
#pragma unroll
    for (int t = 0; t < NT; ++t) acc[t] = (v8f){};

    for (int k0 = 0; k0 < K; k0 += 32) {
        // ---- stage B: lane owns global row k = k0+lane across all NT tiles
        {
            int k = k0 + lane;
            bool kv = (k < K);
            const float* brow = B + (size_t)k * N + n0;
#pragma unroll
            for (int t = 0; t < NT; ++t) {
                _Float16* dst = sB + t * (16 * 32);
#pragma unroll
                for (int q = 0; q < 4; ++q) {
                    float4 v = kv ? ((const float4*)brow)[t * 4 + q]
                                  : make_float4(0.f, 0.f, 0.f, 0.f);
                    int c = q * 4;
                    dst[(c + 0) * 32 + lane] = (_Float16)v.x;
                    dst[(c + 1) * 32 + lane] = (_Float16)v.y;
                    dst[(c + 2) * 32 + lane] = (_Float16)v.z;
                    dst[(c + 3) * 32 + lane] = (_Float16)v.w;
                }
            }
        }
        // ---- A fragment direct from global: two contiguous 8-K runs per lane
        FragU fa;
        {
            const float* p = arow + k0 + hi8;
#pragma unroll
            for (int g = 0; g < 4; ++g) {
                int off = (g < 2) ? g * 4 : 16 + (g - 2) * 4;     // float offset
                bool ok = (k0 + hi8 + off + 4 <= K);               // exact: K % 8 == 0
                float4 v = ok ? *(const float4*)(p + off)
                              : make_float4(0.f, 0.f, 0.f, 0.f);
                fa.u[g * 2 + 0] = pk2(v.x, v.y);
                fa.u[g * 2 + 1] = pk2(v.z, v.w);
            }
        }
        __syncthreads();
#pragma unroll
        for (int t = 0; t < NT; ++t)
            acc[t] = wmma_f16(fa.h, load_b_frag(sB + t * (16 * 32), 32), acc[t]);
        __syncthreads();
    }
    int n = lane & 15, mb = (lane >> 4) << 3;     // C layout: VGPR i -> row mb+i
#pragma unroll
    for (int t = 0; t < NT; ++t)
#pragma unroll
        for (int i = 0; i < 8; ++i) {
            int mm = m0 + mb + i;
            if (mm < M) C[(size_t)mm * N + n0 + t * 16 + n] = acc[t][i];
        }
}

// ---------------- elementwise / attention kernels ----------------
__global__ void k_fill(float* __restrict__ p, float v, int n) {
    int i = blockIdx.x * blockDim.x + threadIdx.x;
    if (i < n) p[i] = v;
}

__global__ void k_alpha(const float* __restrict__ lin, const float* __restrict__ aS,
                        const float* __restrict__ aD, float* __restrict__ asrc,
                        float* __restrict__ adst, int n, int H, int C) {
    int i = blockIdx.x * blockDim.x + threadIdx.x;
    if (i >= n * H) return;
    int node = i / H, h = i % H;
    const float4* hp = (const float4*)(lin + (size_t)node * H * C + h * C);
    const float4* s4 = (const float4*)(aS + h * C);
    const float4* d4 = (const float4*)(aD + h * C);
    float s1 = 0.f, s2 = 0.f;
    for (int c = 0; c < C / 4; ++c) {
        float4 v = hp[c], a = s4[c], b = d4[c];
        s1 += v.x * a.x + v.y * a.y + v.z * a.z + v.w * a.w;
        s2 += v.x * b.x + v.y * b.y + v.z * b.z + v.w * b.w;
    }
    asrc[i] = s1; adst[i] = s2;
}

__global__ void k_edge_max(const int* __restrict__ ei, const float* __restrict__ asrc,
                           const float* __restrict__ adst, float* __restrict__ ebuf,
                           float* __restrict__ mbuf, int E, int N, int H) {
    int i = blockIdx.x * blockDim.x + threadIdx.x;
    int total = (E + N) * H;
    if (i >= total) return;
    int e = i / H, h = i % H;
    int s = (e < E) ? ei[e] : (e - E);
    int d = (e < E) ? ei[E + e] : (e - E);
    float v = asrc[s * H + h] + adst[d * H + h];
    v = (v > 0.f) ? v : 0.2f * v;                 // leaky_relu 0.2
    ebuf[i] = v;
    atomicMaxF(&mbuf[d * H + h], v);
}

__global__ void k_edge_exp(const int* __restrict__ ei, float* __restrict__ ebuf,
                           const float* __restrict__ mbuf, float* __restrict__ sbuf,
                           int E, int N, int H) {
    int i = blockIdx.x * blockDim.x + threadIdx.x;
    int total = (E + N) * H;
    if (i >= total) return;
    int e = i / H, h = i % H;
    int d = (e < E) ? ei[E + e] : (e - E);
    float v = expf(ebuf[i] - mbuf[d * H + h]);
    ebuf[i] = v;
    atomicAdd(&sbuf[d * H + h], v);
}

// one thread per (edge, head, 8-channel group)
__global__ void k_message(const int* __restrict__ ei, const float* __restrict__ lin,
                          const float* __restrict__ ebuf, const float* __restrict__ sbuf,
                          float* __restrict__ agg, int E, int N, int H, int C) {
    const int G = C >> 3;
    long long i = (long long)blockIdx.x * blockDim.x + threadIdx.x;
    long long total = (long long)(E + N) * H * G;
    if (i >= total) return;
    int g  = (int)(i % G);
    int eh = (int)(i / G);
    int h = eh % H, e = eh / H;
    int s = (e < E) ? ei[e] : (e - E);
    int d = (e < E) ? ei[E + e] : (e - E);
    float w = ebuf[eh] / (sbuf[d * H + h] + 1e-16f);
    const float4* lp = (const float4*)(lin + (size_t)s * H * C + h * C + g * 8);
    float* ap = agg + (size_t)d * H * C + h * C + g * 8;
    float4 v0 = lp[0], v1 = lp[1];
    atomicAdd(ap + 0, w * v0.x); atomicAdd(ap + 1, w * v0.y);
    atomicAdd(ap + 2, w * v0.z); atomicAdd(ap + 3, w * v0.w);
    atomicAdd(ap + 4, w * v1.x); atomicAdd(ap + 5, w * v1.y);
    atomicAdd(ap + 6, w * v1.z); atomicAdd(ap + 7, w * v1.w);
}

__global__ void k_bias_act(const float* __restrict__ agg, const float* __restrict__ b,
                           float* __restrict__ out, int n, int F, int act) {
    int i = blockIdx.x * blockDim.x + threadIdx.x;
    if (i >= n * F) return;
    float v = agg[i] + b[i % F];
    if (act) v = (v > 0.f) ? v : (expf(v) - 1.f);  // ELU
    out[i] = v;
}

// ---------------- fused edge MLP: [h_s|h_d|ea](72) ->32 relu ->32 relu ->1 softplus
#define MLP_WAVES 4
__global__ void k_edge_mlp(const float* __restrict__ h3, const int* __restrict__ ei,
                           const float* __restrict__ eattr,
                           const float* __restrict__ Wm1, const float* __restrict__ bm1,
                           const float* __restrict__ Wm2, const float* __restrict__ bm2,
                           const float* __restrict__ Wm3, const float* __restrict__ bm3,
                           float* __restrict__ out, int E) {
    __shared__ _Float16 sW1[32 * 96];                  // transposed [n][k], K padded 72->96
    __shared__ _Float16 sW2[32 * 32];                  // transposed [n][k]
    __shared__ float sB1[32], sB2[32], sW3[32];
    __shared__ _Float16 sA[MLP_WAVES][16 * 96];        // edge-input tiles [row][k]
    __shared__ _Float16 sZ[MLP_WAVES][16 * 32];        // relu(z1) f16
    __shared__ float    sP[MLP_WAVES][16 * 32];        // final partial products
    int tid = threadIdx.x;                             // blockDim.x == 128

    for (int i = tid; i < 32 * 96; i += 128) sW1[i] = (_Float16)0.f;
    __syncthreads();
    for (int i = tid; i < 72 * 32; i += 128) {
        int k = i >> 5, n = i & 31;
        sW1[n * 96 + k] = (_Float16)Wm1[i];
    }
    for (int i = tid; i < 32 * 32; i += 128) {
        int k = i >> 5, n = i & 31;
        sW2[n * 32 + k] = (_Float16)Wm2[i];
    }
    if (tid < 32) { sB1[tid] = bm1[tid]; sB2[tid] = bm2[tid]; sW3[tid] = Wm3[tid]; }
    __syncthreads();

    int wave = tid >> 5, lane = tid & 31;
    int e0 = (blockIdx.x * MLP_WAVES + wave) * 16;
    int r = lane & 15, half = lane >> 4;
    int e = e0 + r;
    bool valid = (e < E);
    int s = 0, d = 0;
    if (valid) { s = ei[e]; d = ei[E + e]; }

    _Float16* a = sA[wave];
    unsigned int* arow = (unsigned int*)(a + r * 96);  // 48 dwords per row
    if (half == 0) {                                   // cols 0..47
        if (valid) {
            const float4* ps = (const float4*)(h3 + (size_t)s * 32);
            const float4* pd = (const float4*)(h3 + (size_t)d * 32);
#pragma unroll
            for (int q = 0; q < 8; ++q) {
                float4 v = ps[q];
                arow[2 * q] = pk2(v.x, v.y); arow[2 * q + 1] = pk2(v.z, v.w);
            }
#pragma unroll
            for (int q = 0; q < 4; ++q) {
                float4 v = pd[q];
                arow[16 + 2 * q] = pk2(v.x, v.y); arow[17 + 2 * q] = pk2(v.z, v.w);
            }
        } else {
            for (int q = 0; q < 24; ++q) arow[q] = 0u;
        }
    } else {                                           // cols 48..95
        if (valid) {
            const float4* pd = (const float4*)(h3 + (size_t)d * 32);
            const float4* pe = (const float4*)(eattr + (size_t)e * 8);
#pragma unroll
            for (int q = 4; q < 8; ++q) {
                float4 v = pd[q];
                arow[16 + 2 * q] = pk2(v.x, v.y); arow[17 + 2 * q] = pk2(v.z, v.w);
            }
#pragma unroll
            for (int q = 0; q < 2; ++q) {
                float4 v = pe[q];
                arow[32 + 2 * q] = pk2(v.x, v.y); arow[33 + 2 * q] = pk2(v.z, v.w);
            }
        } else {
            for (int q = 24; q < 36; ++q) arow[q] = 0u;
        }
        for (int q = 36; q < 48; ++q) arow[q] = 0u;    // K pad 72 -> 96
    }
    __syncthreads();

    // layer 1: z1[16,32] = A[16,96] @ W1[96,32]
    v8f acc0 = {}, acc1 = {};
#pragma unroll
    for (int kc = 0; kc < 3; ++kc) {
        v16h av = load_a_frag(a + kc * 32, 96);
        acc0 = wmma_f16(av, load_b_frag(sW1 + kc * 32, 96), acc0);
        acc1 = wmma_f16(av, load_b_frag(sW1 + 16 * 96 + kc * 32, 96), acc1);
    }
    int n = lane & 15, mb = (lane >> 4) << 3;
    _Float16* z = sZ[wave];
#pragma unroll
    for (int i = 0; i < 8; ++i) {
        z[(mb + i) * 32 + n]      = (_Float16)fmaxf(acc0[i] + sB1[n], 0.f);
        z[(mb + i) * 32 + 16 + n] = (_Float16)fmaxf(acc1[i] + sB1[16 + n], 0.f);
    }
    __syncthreads();

    // layer 2: z2[16,32] = z1[16,32] @ W2[32,32]
    v16h a2 = load_a_frag(z, 32);
    v8f c0 = {}, c1 = {};
    c0 = wmma_f16(a2, load_b_frag(sW2, 32), c0);
    c1 = wmma_f16(a2, load_b_frag(sW2 + 16 * 32, 32), c1);

    float* p = sP[wave];
#pragma unroll
    for (int i = 0; i < 8; ++i) {
        p[(mb + i) * 32 + n]      = fmaxf(c0[i] + sB2[n], 0.f) * sW3[n];
        p[(mb + i) * 32 + 16 + n] = fmaxf(c1[i] + sB2[16 + n], 0.f) * sW3[16 + n];
    }
    __syncthreads();

    if (lane < 16) {                                   // final 32->1 + softplus
        int e2 = e0 + lane;
        if (e2 < E) {
            float sum = bm3[0];
            for (int j = 0; j < 32; ++j) sum += p[lane * 32 + j];
            out[e2] = fmaxf(sum, 0.f) + log1pf(expf(-fabsf(sum)));  // stable softplus
        }
    }
}

// ---------------- host orchestration ----------------
static inline int cdiv(long long a, long long b) { return (int)((a + b - 1) / b); }

template <int NT>
static void gat_layer(const float* in, int Kin, const float* W, const float* aS,
                      const float* aD, const float* b, int H, int C, int act,
                      const int* ei, int E, int N,
                      float* lin, float* agg, float* hout,
                      float* aSrc, float* aDst, float* mbuf, float* sbuf, float* ebuf,
                      hipStream_t stream) {
    int HC = H * C;
    dim3 g(cdiv(N, 16), HC / (16 * NT));
    k_gemm_wmma<NT><<<g, 32, 0, stream>>>(in, W, lin, N, Kin, HC);
    int tNH = N * H;
    k_alpha<<<cdiv(tNH, 256), 256, 0, stream>>>(lin, aS, aD, aSrc, aDst, N, H, C);
    k_fill<<<cdiv(tNH, 256), 256, 0, stream>>>(mbuf, -__builtin_huge_valf(), tNH);
    k_fill<<<cdiv(tNH, 256), 256, 0, stream>>>(sbuf, 0.f, tNH);
    k_fill<<<cdiv((long long)N * HC, 256), 256, 0, stream>>>(agg, 0.f, N * HC);
    int Et = E + N;
    k_edge_max<<<cdiv((long long)Et * H, 256), 256, 0, stream>>>(ei, aSrc, aDst, ebuf, mbuf, E, N, H);
    k_edge_exp<<<cdiv((long long)Et * H, 256), 256, 0, stream>>>(ei, ebuf, mbuf, sbuf, E, N, H);
    k_message<<<cdiv((long long)Et * H * (C >> 3), 256), 256, 0, stream>>>(ei, lin, ebuf, sbuf, agg, E, N, H, C);
    k_bias_act<<<cdiv((long long)N * HC, 256), 256, 0, stream>>>(agg, b, hout, N, HC, act);
}

extern "C" void kernel_launch(void* const* d_in, const int* in_sizes, int n_in,
                              void* d_out, int out_size, void* d_ws, size_t ws_size,
                              hipStream_t stream) {
    (void)n_in; (void)out_size; (void)ws_size;
    const float* x     = (const float*)d_in[0];
    const int*   ei    = (const int*)d_in[1];
    const float* eattr = (const float*)d_in[2];
    const float* W1 = (const float*)d_in[3];  const float* aS1 = (const float*)d_in[4];
    const float* aD1 = (const float*)d_in[5]; const float* b1  = (const float*)d_in[6];
    const float* W2 = (const float*)d_in[7];  const float* aS2 = (const float*)d_in[8];
    const float* aD2 = (const float*)d_in[9]; const float* b2  = (const float*)d_in[10];
    const float* W3 = (const float*)d_in[11]; const float* aS3 = (const float*)d_in[12];
    const float* aD3 = (const float*)d_in[13]; const float* b3 = (const float*)d_in[14];
    const float* Wm1 = (const float*)d_in[15]; const float* bm1 = (const float*)d_in[16];
    const float* Wm2 = (const float*)d_in[17]; const float* bm2 = (const float*)d_in[18];
    const float* Wm3 = (const float*)d_in[19]; const float* bm3 = (const float*)d_in[20];

    int N  = in_sizes[0] / 16;     // NF = 16
    int E  = in_sizes[2] / 8;      // EF = 8

    float* ws   = (float*)d_ws;
    float* bufA = ws;                               // [N,128] lin
    float* bufB = bufA + (size_t)N * 128;           // [N,128] agg / h
    float* bufC = bufB + (size_t)N * 128;           // [N,32]  layer-3 out
    float* aSrc = bufC + (size_t)N * 32;            // [N,4]
    float* aDst = aSrc + (size_t)N * 4;
    float* mbuf = aDst + (size_t)N * 4;
    float* sbuf = mbuf + (size_t)N * 4;
    float* ebuf = sbuf + (size_t)N * 4;             // [E+N,4]

    // layer 1: x[N,16] -> h1 = bufB
    gat_layer<4>(x, 16, W1, aS1, aD1, b1, 4, 32, 1, ei, E, N,
                 bufA, bufB, bufB, aSrc, aDst, mbuf, sbuf, ebuf, stream);
    // layer 2: h1[N,128] -> h2 = bufB (lin in bufA; h1 dead after GEMM)
    gat_layer<4>(bufB, 128, W2, aS2, aD2, b2, 4, 32, 1, ei, E, N,
                 bufA, bufB, bufB, aSrc, aDst, mbuf, sbuf, ebuf, stream);
    // layer 3: h2[N,128] -> h3 = bufC (H=1, no activation)
    gat_layer<2>(bufB, 128, W3, aS3, aD3, b3, 1, 32, 0, ei, E, N,
                 bufA, bufC, bufC, aSrc, aDst, mbuf, sbuf, ebuf, stream);

    // fused edge MLP over original E edges
    k_edge_mlp<<<cdiv(E, 16 * MLP_WAVES), 128, 0, stream>>>(
        bufC, ei, eattr, Wm1, bm1, Wm2, bm2, Wm3, bm3, (float*)d_out, E);
}
